// PointNetSetAbstractionMsg_29798483100269
// MI455X (gfx1250) — matrix operand order repaired and checked by reading, not compiled, not run on hardware
//
#include <hip/hip_runtime.h>
#include <hip/hip_bf16.h>
#include <math.h>

typedef _Float16 f16;
typedef __attribute__((ext_vector_type(16))) _Float16 v16h;
typedef __attribute__((ext_vector_type(8)))  _Float16 v8h;
typedef __attribute__((ext_vector_type(8)))  float    v8f;

#define Bb   8
#define Nn   8192
#define Ss   1024
#define KMAX 128

// ---------------------------------------------------------------------------
// 1) new_xyz output: xyz[:, :, :1024] copied to d_out (B,3,S) flat.
// ---------------------------------------------------------------------------
__global__ __launch_bounds__(256) void copy_newxyz(const float* __restrict__ xyz,
                                                   float* __restrict__ out) {
  int i = blockIdx.x * 256 + threadIdx.x;
  if (i >= Bb * 3 * Ss) return;
  int b = i / (3 * Ss);
  int r = i - b * 3 * Ss;
  int c = r >> 10;
  int s = r & (Ss - 1);
  out[i] = xyz[(long long)b * 3 * Nn + c * Nn + s];
}

// ---------------------------------------------------------------------------
// 2) top-128 nearest neighbors per query (sorted ascending, ties by lower
//    index — matches jax.lax.top_k on -d). One block per (b,s); scales
//    K=16/32 use prefixes of the sorted 128.
// ---------------------------------------------------------------------------
__global__ __launch_bounds__(256) void topk_kernel(const float* __restrict__ xyz,
                                                   int* __restrict__ idxOut) {
  __shared__ unsigned int du[Nn];
  __shared__ unsigned long long wmin[8];
  __shared__ unsigned long long winner;

  int b = blockIdx.x >> 10;
  int s = blockIdx.x & (Ss - 1);
  const float* base = xyz + (long long)b * 3 * Nn;
  float qx = base[s], qy = base[Nn + s], qz = base[2 * Nn + s];
  float qq = qx * qx + qy * qy + qz * qz;

  int tid = threadIdx.x;
  for (int j = 0; j < Nn / 256; ++j) {
    int n = tid + j * 256;
    float px = base[n], py = base[Nn + n], pz = base[2 * Nn + n];
    float pp = px * px + py * py + pz * pz;
    float dt = qx * px + qy * py + qz * pz;
    float d = qq + pp - 2.0f * dt;
    unsigned int u = __float_as_uint(d);
    u = (u & 0x80000000u) ? ~u : (u | 0x80000000u);
    du[n] = u;
  }
  __syncthreads();

  int nb = tid * 32;
  unsigned long long cur = ~0ULL;
  for (int j = 0; j < 32; ++j) {
    unsigned long long key =
        ((unsigned long long)du[nb + j] << 32) | (unsigned int)(nb + j);
    cur = (cur < key) ? cur : key;
  }

  int lane = tid & 31, wave = tid >> 5;
  int* out = idxOut + (long long)blockIdx.x * KMAX;

  for (int it = 0; it < KMAX; ++it) {
    unsigned long long k0 = cur;
    #pragma unroll
    for (int off = 16; off >= 1; off >>= 1) {
      unsigned long long o = __shfl_xor(k0, off);
      k0 = (k0 < o) ? k0 : o;
    }
    if (lane == 0) wmin[wave] = k0;
    __syncthreads();
    if (tid == 0) {
      unsigned long long mm = wmin[0];
      #pragma unroll
      for (int w = 1; w < 8; ++w) mm = (mm < wmin[w]) ? mm : wmin[w];
      winner = mm;
    }
    __syncthreads();
    unsigned long long w = winner;
    int n = (int)(w & 0xffffffffULL);
    if (tid == 0) out[it] = n;
    if (tid == (n >> 5)) {
      du[n] = 0xFFFFFFFFu;
      cur = ~0ULL;
      for (int j = 0; j < 32; ++j) {
        unsigned long long key =
            ((unsigned long long)du[nb + j] << 32) | (unsigned int)(nb + j);
        cur = (cur < key) ? cur : key;
      }
    }
    __syncthreads();
  }
}

// ---------------------------------------------------------------------------
// 3) gather: X0[m][32] f16 (channels 6..31 zero-padded), m = ((b*S+s)*K + k).
//    channels 0-2 = points[idx], 3-5 = xyz[idx] - xyz[query]
// ---------------------------------------------------------------------------
__global__ __launch_bounds__(256) void gather_kernel(const float* __restrict__ xyz,
                                                     const float* __restrict__ pts,
                                                     const int* __restrict__ idx,
                                                     f16* __restrict__ X0,
                                                     int K, long long M) {
  long long m = (long long)blockIdx.x * 256 + threadIdx.x;
  if (m >= M) return;
  int k = (int)(m % K);
  long long bs = m / K;
  int s = (int)(bs & (Ss - 1));
  int b = (int)(bs >> 10);
  int p = idx[bs * KMAX + k];
  long long xb = (long long)b * 3 * Nn;

  v8h v0;
  #pragma unroll
  for (int h = 0; h < 8; ++h) v0[h] = (f16)0.0f;
  v8h vz = v0;
  #pragma unroll
  for (int c = 0; c < 3; ++c) {
    v0[c]     = (f16)pts[xb + c * Nn + p];
    v0[3 + c] = (f16)(xyz[xb + c * Nn + p] - xyz[xb + c * Nn + s]);
  }
  v8h* dst = (v8h*)(X0 + m * 32);
  dst[0] = v0; dst[1] = vz; dst[2] = vz; dst[3] = vz;
}

// ---------------------------------------------------------------------------
// 4) GEMM (1x1 conv) via v_wmma_f32_16x16x32_f16.
//    Activations are [m][CPAD] f16 (CPAD = NCHUNK*32 = ldIn = cIn or padded).
//    Lazy BN+ReLU of the previous layer applied on the B-operand load
//    (scale/shift cached in LDS). Bias omitted: it cancels in batch-norm.
//    Per wave: one 16-point column, loop over o-tiles; A from LDS (b128),
//    B via 2x global b128 per chunk, Y stored as one b128 per o-tile.
// ---------------------------------------------------------------------------
template <int NCHUNK>
__global__ __launch_bounds__(256) void gemm_wmma(
    const f16* __restrict__ Xin, f16* __restrict__ Yout,
    const float* __restrict__ W,
    const float* __restrict__ inScale, const float* __restrict__ inShift,
    int cIn, int cOut, long long M) {
  const int CPAD = NCHUNK * 32;
  __shared__ f16 wlds[128 * 96];           // cOut * CPAD <= 12288
  __shared__ float ssc[128], ssh[128];

  int tid = threadIdx.x;
  for (int e = tid; e < cOut * CPAD; e += 256) {
    int o = e / CPAD, c = e - o * CPAD;
    wlds[e] = (c < cIn) ? (f16)W[o * cIn + c] : (f16)0.0f;
  }
  if (inScale && tid < cIn) { ssc[tid] = inScale[tid]; ssh[tid] = inShift[tid]; }
  __syncthreads();

  int lane = tid & 31, wave = tid >> 5;
  long long m0 = ((long long)blockIdx.x * 8 + wave) * 16;
  int nl    = lane & 15;
  int koff  = (lane < 16) ? 0 : 16;        // B fragment: K-half per lane group
  int rbase = (lane < 16) ? 0 : 8;         // C: M offset per lane group
  long long mcol = m0 + nl;
  const f16* xrow = Xin + mcol * CPAD;
  __builtin_prefetch(xrow + (long long)128 * CPAD, 0, 1);

  v16h bfrag[NCHUNK];
  #pragma unroll
  for (int ch = 0; ch < NCHUNK; ++ch) {
    v8h lo = *(const v8h*)(xrow + ch * 32 + koff);
    v8h hi = *(const v8h*)(xrow + ch * 32 + koff + 8);
    v16h bb;
    if (inScale) {                         // lazy BN + ReLU of previous layer
      #pragma unroll
      for (int h = 0; h < 8; ++h) {
        int c = ch * 32 + koff + h;
        float v1 = fmaf((float)lo[h], ssc[c], ssh[c]);
        float v2 = fmaf((float)hi[h], ssc[c + 8], ssh[c + 8]);
        bb[h]     = (f16)(v1 > 0.0f ? v1 : 0.0f);
        bb[h + 8] = (f16)(v2 > 0.0f ? v2 : 0.0f);
      }
    } else {
      #pragma unroll
      for (int h = 0; h < 8; ++h) { bb[h] = lo[h]; bb[h + 8] = hi[h]; }
    }
    bfrag[ch] = bb;
  }

  for (int ot = 0; ot < cOut / 16; ++ot) {
    v8f acc = {};
    #pragma unroll
    for (int ch = 0; ch < NCHUNK; ++ch) {
      const f16* wr = &wlds[(ot * 16 + nl) * CPAD + ch * 32];
      v8h alo = *(const v8h*)(wr + koff);
      v8h ahi = *(const v8h*)(wr + koff + 8);
      v16h aa;
      #pragma unroll
      for (int h = 0; h < 8; ++h) { aa[h] = alo[h]; aa[h + 8] = ahi[h]; }
      acc = __builtin_amdgcn_wmma_f32_16x16x32_f16(
          false, aa, false, bfrag[ch], (short)0, acc, false, false);
    }
    v8h ys;
    #pragma unroll
    for (int r = 0; r < 8; ++r) ys[r] = (f16)acc[r];
    *(v8h*)(Yout + mcol * cOut + ot * 16 + rbase) = ys;
  }
}

// ---------------------------------------------------------------------------
// 5) per-channel sum / sum-of-squares over Y[m][cOut] (memory-shaped pass).
//    Each thread owns 8 consecutive channels, b128 loads, register accum.
// ---------------------------------------------------------------------------
__global__ void stats_kernel(const f16* __restrict__ Y,
                             float* __restrict__ gSum, float* __restrict__ gSumSq,
                             int cOut, long long mPerBlock, int xdim, int ydim) {
  __shared__ float lsum[128], lsq[128];
  int tid = threadIdx.x;
  if (tid < cOut) { lsum[tid] = 0.0f; lsq[tid] = 0.0f; }
  __syncthreads();

  int tx = tid % xdim, ty = tid / xdim;
  long long m0 = (long long)blockIdx.x * mPerBlock;
  float s[8], q[8];
  #pragma unroll
  for (int r = 0; r < 8; ++r) { s[r] = 0.0f; q[r] = 0.0f; }
  for (long long m = m0 + ty; m < m0 + mPerBlock; m += ydim) {
    v8h v = *(const v8h*)(Y + m * cOut + tx * 8);
    #pragma unroll
    for (int r = 0; r < 8; ++r) {
      float f = (float)v[r];
      s[r] += f;
      q[r] += f * f;
    }
  }
  #pragma unroll
  for (int r = 0; r < 8; ++r) {
    atomicAdd(&lsum[tx * 8 + r], s[r]);
    atomicAdd(&lsq[tx * 8 + r], q[r]);
  }
  __syncthreads();
  if (tid < cOut) {
    atomicAdd(&gSum[tid], lsum[tid]);
    atomicAdd(&gSumSq[tid], lsq[tid]);
  }
}

// ---------------------------------------------------------------------------
// 6) batchnorm finalize: scale = g*rsqrt(var+eps), shift = beta - mean*scale
// ---------------------------------------------------------------------------
__global__ void bn_finalize(const float* __restrict__ gSum,
                            const float* __restrict__ gSumSq,
                            const float* __restrict__ g,
                            const float* __restrict__ beta,
                            float* __restrict__ outScale,
                            float* __restrict__ outShift,
                            int cOut, float invM) {
  int c = threadIdx.x;
  if (c < cOut) {
    float mean = gSum[c] * invM;
    float var  = gSumSq[c] * invM - mean * mean;
    float sc   = g[c] * rsqrtf(var + 1e-5f);
    outScale[c] = sc;
    outShift[c] = beta[c] - mean * sc;
  }
}

// ---------------------------------------------------------------------------
// 7) last-layer BN+ReLU and max over K; 8 channels per thread, b128 loads.
// ---------------------------------------------------------------------------
__global__ __launch_bounds__(256) void bn_relu_maxpool(
    const f16* __restrict__ Y, const float* __restrict__ scale,
    const float* __restrict__ shift, float* __restrict__ out2,
    int K, int cOut, int cBase, int xdim) {
  long long t = (long long)blockIdx.x * 256 + threadIdx.x;
  long long total = (long long)Bb * Ss * xdim;
  if (t >= total) return;
  int g8 = (int)(t % xdim);
  long long bs = t / xdim;
  int s = (int)(bs & (Ss - 1));
  int b = (int)(bs >> 10);

  float sc[8], sh[8], mx[8];
  #pragma unroll
  for (int r = 0; r < 8; ++r) {
    sc[r] = scale[g8 * 8 + r];
    sh[r] = shift[g8 * 8 + r];
    mx[r] = 0.0f;                           // ReLU output >= 0
  }
  const f16* yp = Y + (bs * K) * cOut + g8 * 8;
  for (int k = 0; k < K; ++k) {
    v8h v = *(const v8h*)(yp + (long long)k * cOut);
    #pragma unroll
    for (int r = 0; r < 8; ++r) {
      float f = fmaf((float)v[r], sc[r], sh[r]);
      f = f > 0.0f ? f : 0.0f;
      mx[r] = fmaxf(mx[r], f);
    }
  }
  #pragma unroll
  for (int r = 0; r < 8; ++r)
    out2[((long long)b * 320 + (cBase + g8 * 8 + r)) * Ss + s] = mx[r];
}

// ---------------------------------------------------------------------------
// host-side launch
// ---------------------------------------------------------------------------
static void launch_gemm(int nchunk, dim3 g, dim3 bl, hipStream_t st,
                        const f16* Xin, f16* Yout, const float* W,
                        const float* inSc, const float* inSh,
                        int cIn, int cOut, long long M) {
  if (nchunk == 1)
    gemm_wmma<1><<<g, bl, 0, st>>>(Xin, Yout, W, inSc, inSh, cIn, cOut, M);
  else if (nchunk == 2)
    gemm_wmma<2><<<g, bl, 0, st>>>(Xin, Yout, W, inSc, inSh, cIn, cOut, M);
  else
    gemm_wmma<3><<<g, bl, 0, st>>>(Xin, Yout, W, inSc, inSh, cIn, cOut, M);
}

extern "C" void kernel_launch(void* const* d_in, const int* in_sizes, int n_in,
                              void* d_out, int out_size, void* d_ws, size_t ws_size,
                              hipStream_t stream) {
  const float* xyz = (const float*)d_in[0];
  const float* pts = (const float*)d_in[1];

  static const int K_LIST[3] = {16, 32, 128};
  static const int CBASE[3]  = {0, 64, 192};
  static const int DIMS[3][3][2] = {
      {{6, 32}, {32, 32}, {32, 64}},
      {{6, 64}, {64, 64}, {64, 128}},
      {{6, 64}, {64, 96}, {96, 128}}};

  float* out1 = (float*)d_out;                 // (8,3,1024)
  float* out2 = (float*)d_out + Bb * 3 * Ss;   // (8,320,1024)

  char* ws = (char*)d_ws;
  size_t off = 0;
  auto carve = [&](size_t bytes) {
    size_t o = off;
    off = (off + bytes + 255) & ~(size_t)255;
    return o;
  };
  int*   idxb   = (int*)  (ws + carve((size_t)Bb * Ss * KMAX * 4));
  float* gSum   = (float*)(ws + carve(256 * 4));
  float* gSumSq = gSum + 128;
  float* scA    = (float*)(ws + carve(128 * 4));
  float* shA    = (float*)(ws + carve(128 * 4));
  float* scB    = (float*)(ws + carve(128 * 4));
  float* shB    = (float*)(ws + carve(128 * 4));
  f16*   bufA   = (f16*)  (ws + carve((size_t)96  * 1048576 * 2));  // X0(pad32)/Y1
  f16*   bufB   = (f16*)  (ws + carve((size_t)128 * 1048576 * 2));  // Y0/Y2

  copy_newxyz<<<(Bb * 3 * Ss + 255) / 256, 256, 0, stream>>>(xyz, out1);
  topk_kernel<<<Bb * Ss, 256, 0, stream>>>(xyz, idxb);

  for (int i = 0; i < 3; ++i) {
    int K = K_LIST[i];
    long long M = (long long)Bb * Ss * K;

    gather_kernel<<<(unsigned)(M / 256), 256, 0, stream>>>(xyz, pts, idxb, bufA, K, M);

    const f16* Xin = bufA;
    f16* Yout = bufB;
    const float* inSc = nullptr;
    const float* inSh = nullptr;
    float* oSc = scA;
    float* oSh = shA;

    for (int j = 0; j < 3; ++j) {
      int cIn = DIMS[i][j][0], cOut = DIMS[i][j][1];
      int base = 2 + i * 12 + j * 4;
      const float* W    = (const float*)d_in[base + 0];
      const float* Gam  = (const float*)d_in[base + 2];
      const float* Beta = (const float*)d_in[base + 3];

      int nchunk = (cIn + 31) / 32;
      launch_gemm(nchunk, dim3((unsigned)(M / 128)), dim3(256), stream,
                  Xin, Yout, W, inSc, inSh, cIn, cOut, M);

      hipMemsetAsync(gSum, 0, 256 * 4, stream);
      int xdim = cOut / 8;
      int ydim = 256 / xdim;
      stats_kernel<<<256, xdim * ydim, 0, stream>>>(Yout, gSum, gSumSq, cOut,
                                                    M / 256, xdim, ydim);
      bn_finalize<<<1, 128, 0, stream>>>(gSum, gSumSq, Gam, Beta, oSc, oSh,
                                         cOut, 1.0f / (float)M);
      Xin  = Yout;
      Yout = (Yout == bufB) ? bufA : bufB;
      inSc = oSc; inSh = oSh;
      if (oSc == scA) { oSc = scB; oSh = shB; } else { oSc = scA; oSh = shA; }
    }

    int oLast = DIMS[i][2][1];
    int xdim = oLast / 8;
    long long total = (long long)Bb * Ss * xdim;
    bn_relu_maxpool<<<(unsigned)((total + 255) / 256), 256, 0, stream>>>(
        Xin, inSc, inSh, out2, K, oLast, CBASE[i], xdim);
  }
}
